// PointNet2Backbone_26456998543766
// MI455X (gfx1250) — compile-verified, matrix operand order checked by reading, not run on hardware
//
#include <hip/hip_runtime.h>

typedef _Float16 f16;
typedef __attribute__((ext_vector_type(16))) _Float16     v16h;
typedef __attribute__((ext_vector_type(8)))  float        v8f;
typedef __attribute__((ext_vector_type(4)))  unsigned int u32x4;
typedef __attribute__((ext_vector_type(8)))  int          i32x8;
typedef __attribute__((ext_vector_type(4)))  int          i32x4;

// ---------------------------------------------------------------------------
// (B,3,N) -> (B,N,3)
// ---------------------------------------------------------------------------
__global__ void transpose_xyz_kernel(const float* __restrict__ x,
                                     float* __restrict__ xyz, int B, int N) {
    int g = blockIdx.x * blockDim.x + threadIdx.x;
    if (g >= B * N) return;
    int b = g / N, n = g % N;
    xyz[(size_t)g * 3 + 0] = x[((size_t)b * 3 + 0) * N + n];
    xyz[(size_t)g * 3 + 1] = x[((size_t)b * 3 + 1) * N + n];
    xyz[(size_t)g * 3 + 2] = x[((size_t)b * 3 + 2) * N + n];
}

// ---------------------------------------------------------------------------
// Farthest point sampling: one workgroup (256 threads, 8 waves) per batch.
// Points live in registers; only the 256-wide argmax reduction uses LDS.
// ---------------------------------------------------------------------------
template<int PT>
__global__ void fps_kernel(const float* __restrict__ xyz,
                           float* __restrict__ new_xyz, int N, int npoint) {
    const int b = blockIdx.x, tid = threadIdx.x;
    float px[PT], py[PT], pz[PT], dv[PT];
    const float* base = xyz + (size_t)b * N * 3;
#pragma unroll
    for (int t = 0; t < PT; ++t) {
        int i = tid + t * 256;
        px[t] = base[i * 3 + 0];
        py[t] = base[i * 3 + 1];
        pz[t] = base[i * 3 + 2];
        dv[t] = 1e10f;
    }
    __shared__ float s_cx, s_cy, s_cz;
    __shared__ float s_best[256];
    __shared__ int   s_bidx[256];
    if (tid == 0) { s_cx = px[0]; s_cy = py[0]; s_cz = pz[0]; }
    __syncthreads();
    for (int it = 0; it < npoint; ++it) {
        float cx = s_cx, cy = s_cy, cz = s_cz;
        if (tid == 0) {
            float* o = new_xyz + ((size_t)b * npoint + it) * 3;
            o[0] = cx; o[1] = cy; o[2] = cz;
        }
        float best = -1.0f; int bidx = 0;
#pragma unroll
        for (int t = 0; t < PT; ++t) {
            float dx = px[t] - cx, dy = py[t] - cy, dz = pz[t] - cz;
            float d = dx * dx + dy * dy + dz * dz;
            dv[t] = fminf(dv[t], d);
            if (dv[t] > best) { best = dv[t]; bidx = tid + t * 256; }
        }
        s_best[tid] = best; s_bidx[tid] = bidx;
        __syncthreads();
        for (int off = 128; off > 0; off >>= 1) {
            if (tid < off) {
                float ov = s_best[tid + off]; int oi = s_bidx[tid + off];
                if (ov > s_best[tid] || (ov == s_best[tid] && oi < s_bidx[tid])) {
                    s_best[tid] = ov; s_bidx[tid] = oi;
                }
            }
            __syncthreads();
        }
        int w = s_bidx[0];
        if ((w & 255) == tid) {              // owner publishes new centroid
            int t = w >> 8;
            s_cx = px[t]; s_cy = py[t]; s_cz = pz[t];
        }
        __syncthreads();
    }
}

// ---------------------------------------------------------------------------
// Ball query: first K in-radius indices ascending, padded with first hit.
// ---------------------------------------------------------------------------
__global__ void ball_query_kernel(const float* __restrict__ xyz,
                                  const float* __restrict__ nxyz,
                                  int* __restrict__ idx,
                                  int B, int N, int S, int K, float r2) {
    int g = blockIdx.x * blockDim.x + threadIdx.x;
    if (g >= B * S) return;
    int b = g / S;
    const float* base = xyz + (size_t)b * N * 3;
    float cx = nxyz[(size_t)g * 3 + 0];
    float cy = nxyz[(size_t)g * 3 + 1];
    float cz = nxyz[(size_t)g * 3 + 2];
    int* out = idx + (size_t)g * K;
    int cnt = 0, first = -1;
    for (int i = 0; i < N && cnt < K; ++i) {
        float dx = base[i * 3 + 0] - cx;
        float dy = base[i * 3 + 1] - cy;
        float dz = base[i * 3 + 2] - cz;
        if (dx * dx + dy * dy + dz * dz <= r2) {
            if (first < 0) first = i;
            out[cnt++] = i;
        }
    }
    if (first < 0) first = 0;
    for (; cnt < K; ++cnt) out[cnt] = first;
}

// ---------------------------------------------------------------------------
// Grouped feature rows (M = B*S*K) x Cpad in f16:
// [xyz - center | gathered prev features | zero pad]
// ---------------------------------------------------------------------------
__global__ void group_kernel(const float* __restrict__ xyz,
                             const float* __restrict__ nxyz,
                             const int* __restrict__ idx,
                             const float* __restrict__ feats,
                             f16* __restrict__ out,
                             int B, int N, int S, int K, int C, int Cpad) {
    int g = blockIdx.x * blockDim.x + threadIdx.x;
    int M = B * S * K;
    if (g >= M) return;
    int bs = g / K;
    int b = bs / S;
    int pi = idx[g];
    const float* p = xyz + ((size_t)b * N + pi) * 3;
    const float* c = nxyz + (size_t)bs * 3;
    f16* o = out + (size_t)g * Cpad;
    o[0] = (f16)(p[0] - c[0]);
    o[1] = (f16)(p[1] - c[1]);
    o[2] = (f16)(p[2] - c[2]);
    int j = 3;
    if (feats) {
        const float* f = feats + ((size_t)b * N + pi) * C;
        for (int cc = 0; cc < C; ++cc) o[j++] = (f16)f[cc];
    }
    for (; j < Cpad; ++j) o[j] = (f16)0.0f;
}

// group_all: rows = B*N, no centering, [xyz | feats | pad]
__global__ void group_all_kernel(const float* __restrict__ xyz,
                                 const float* __restrict__ feats,
                                 f16* __restrict__ out,
                                 int B, int N, int C, int Cpad) {
    int g = blockIdx.x * blockDim.x + threadIdx.x;
    if (g >= B * N) return;
    const float* p = xyz + (size_t)g * 3;
    const float* f = feats + (size_t)g * C;
    f16* o = out + (size_t)g * Cpad;
    o[0] = (f16)p[0]; o[1] = (f16)p[1]; o[2] = (f16)p[2];
    int j = 3;
    for (int cc = 0; cc < C; ++cc) o[j++] = (f16)f[cc];
    for (; j < Cpad; ++j) o[j] = (f16)0.0f;
}

// ---------------------------------------------------------------------------
// Fold conv bias + BN into per-channel scale/bias; W (O,C) f32 -> (O,Cpad) f16.
// ---------------------------------------------------------------------------
__global__ void prep_weights_kernel(const float* __restrict__ W,
                                    const float* __restrict__ bb,
                                    const float* __restrict__ gamma,
                                    const float* __restrict__ beta,
                                    const float* __restrict__ mean,
                                    const float* __restrict__ var,
                                    f16* __restrict__ Wf,
                                    float* __restrict__ scale,
                                    float* __restrict__ bias,
                                    int O, int C, int Cpad) {
    int o = blockIdx.x * blockDim.x + threadIdx.x;
    if (o >= O) return;
    float s = gamma[o] * rsqrtf(var[o] + 1e-5f);
    scale[o] = s;
    bias[o]  = (bb[o] - mean[o]) * s + beta[o];
    const float* wr = W + (size_t)o * C;
    f16* wo = Wf + (size_t)o * Cpad;
    int c = 0;
    for (; c < C; ++c) wo[c] = (f16)wr[c];
    for (; c < Cpad; ++c) wo[c] = (f16)0.0f;
}

// ---------------------------------------------------------------------------
// WMMA GEMM + fused BN/ReLU epilogue.
//   Y(M,O) = relu( X(M,Kp) @ W(O,Kp)^T * scale + bias )
// 256 threads = 8 waves. Block covers a 128(M) x 64(N) output tile; each wave
// owns 16x64 (4 f32 accumulators -> 4 independent back-to-back WMMAs/K-step,
// 4x less A-matrix re-read traffic than a 16x16-per-wave mapping).
// The 64-row weight strip (<= 64*288 f16 = 36 KB) is DMA'd into LDS once per
// block by wave 0 via the Tensor Data Mover (2-D D# descriptor, data_size=2B,
// tile Kp x 64, stride Kp), synchronized with s_wait_tensorcnt + barrier.
// Fragment layouts per ISA 7.12.2:
//   A (16-bit 16x32): lane L, elem j -> k = (j/8)*16 + (L/16)*8 + (j%8), m=L%16
//   B mirrors A with n = L%16;  C/D f32: elem r -> M = r + 8*(L/16), N = L%16.
// ---------------------------------------------------------------------------
__global__ void mlp_wmma_kernel(const f16* __restrict__ X,
                                const f16* __restrict__ Wt,
                                const float* __restrict__ scale,
                                const float* __restrict__ bias,
                                f16* __restrict__ Y,
                                int M, int Kp, int O) {
    __shared__ f16 sW[64 * 288];

    int wave = threadIdx.x >> 5;
    int lane = threadIdx.x & 31;
    int m0 = (blockIdx.x * 8 + wave) * 16;
    int n0 = blockIdx.y * 64;
    int half = lane >> 4;      // 0 / 1
    int lr   = lane & 15;      // row of A / col of B

    // ---- TDM: stage 64 x Kp f16 weight strip into LDS (wave 0 only) ----
    if (threadIdx.x < 32) {
        unsigned long long ga =
            (unsigned long long)(uintptr_t)(Wt + (size_t)n0 * Kp);
        unsigned int ldsoff = (unsigned int)(uintptr_t)(&sW[0]); // LDS offset = addr[31:0]
        u32x4 g0;
        g0[0] = 1u;                                   // count=1, user descriptor
        g0[1] = ldsoff;                               // lds_addr
        g0[2] = (unsigned int)(ga & 0xffffffffu);     // global_addr[31:0]
        g0[3] = (unsigned int)((ga >> 32) & 0x01ffffffu) | (2u << 30); // [56:32] | type=2
        i32x8 g1;
        g1[0] = (int)(1u << 16);                      // data_size = 2 bytes
        g1[1] = (int)((unsigned)Kp << 16);            // tensor_dim0[15:0]
        g1[2] = (int)(((unsigned)Kp >> 16) | (64u << 16)); // dim0 hi | tensor_dim1=64
        g1[3] = (int)((unsigned)Kp << 16);            // dim1 hi=0 | tile_dim0=Kp
        g1[4] = 64;                                   // tile_dim1=64, tile_dim2=0
        g1[5] = Kp;                                   // tensor_dim0_stride lo
        g1[6] = 0;                                    // stride0 hi | stride1 lo
        g1[7] = 0;
        i32x4 z4 = {0, 0, 0, 0};
        i32x8 z8 = {0, 0, 0, 0, 0, 0, 0, 0};
        __builtin_amdgcn_tensor_load_to_lds(g0, g1, z4, z4, z8, 0);
        __builtin_amdgcn_s_wait_tensorcnt(0);
    }
    __syncthreads();

    const f16* xrow = X + (size_t)(m0 + lr) * Kp;
    int wb0 = (0 * 16 + lr) * Kp;
    int wb1 = (1 * 16 + lr) * Kp;
    int wb2 = (2 * 16 + lr) * Kp;
    int wb3 = (3 * 16 + lr) * Kp;

    v8f acc0 = {}, acc1 = {}, acc2 = {}, acc3 = {};
    for (int kk = 0; kk < Kp; kk += 32) {
        __builtin_prefetch(xrow + kk + 32, 0, 0);  // global_prefetch_b8
        v16h a, b0, b1, b2, b3;
#pragma unroll
        for (int j = 0; j < 16; ++j) {
            int k = kk + ((j >> 3) << 4) + (half << 3) + (j & 7);
            a[j]  = xrow[k];
            b0[j] = sW[wb0 + k];
            b1[j] = sW[wb1 + k];
            b2[j] = sW[wb2 + k];
            b3[j] = sW[wb3 + k];
        }
        acc0 = __builtin_amdgcn_wmma_f32_16x16x32_f16(false, a, false, b0,
                                                      (short)0, acc0, false, false);
        acc1 = __builtin_amdgcn_wmma_f32_16x16x32_f16(false, a, false, b1,
                                                      (short)0, acc1, false, false);
        acc2 = __builtin_amdgcn_wmma_f32_16x16x32_f16(false, a, false, b2,
                                                      (short)0, acc2, false, false);
        acc3 = __builtin_amdgcn_wmma_f32_16x16x32_f16(false, a, false, b3,
                                                      (short)0, acc3, false, false);
    }

#pragma unroll
    for (int t = 0; t < 4; ++t) {
        v8f acc = (t == 0) ? acc0 : (t == 1) ? acc1 : (t == 2) ? acc2 : acc3;
        int n = n0 + t * 16 + lr;
        float sc = scale[n];
        float bi = bias[n];
#pragma unroll
        for (int r = 0; r < 8; ++r) {
            int m = m0 + r + half * 8;
            float v = acc[r] * sc + bi;
            v = v > 0.0f ? v : 0.0f;
            Y[(size_t)m * O + n] = (f16)v;
        }
    }
}

// ---------------------------------------------------------------------------
// Max-pool over the K neighbors of each group; f16 activations -> f32 out.
// ---------------------------------------------------------------------------
__global__ void maxpool_kernel(const f16* __restrict__ act,
                               float* __restrict__ out,
                               int BS, int K, int O) {
    int g = blockIdx.x * blockDim.x + threadIdx.x;
    if (g >= BS * O) return;
    int bs = g / O, c = g % O;
    float v = -3.4e38f;
    for (int k = 0; k < K; ++k) {
        float a = (float)act[((size_t)bs * K + k) * O + c];
        v = a > v ? a : v;
    }
    out[g] = v;
}

// ---------------------------------------------------------------------------
// Host orchestration
// ---------------------------------------------------------------------------
extern "C" void kernel_launch(void* const* d_in, const int* in_sizes, int n_in,
                              void* d_out, int out_size, void* d_ws, size_t ws_size,
                              hipStream_t stream) {
    const int B = 8, N1 = 4096, S1 = 1024, K1 = 32;
    const int S2 = 512, K2 = 32, S3 = 128, K3 = 64, N4 = 128;

    const float* x = (const float*)d_in[0];
    auto P = [&](int L, int j) { return (const float*)d_in[1 + L * 6 + j]; };

    // per-layer dims (12 layers): O, C, Cpad (pad to multiple of 32)
    const int LO[12]  = { 64, 64,128, 128,128,256, 256,256,256, 256,256,256 };
    const int LC[12]  = {  3, 64, 64, 131,128,128, 259,256,256, 259,256,256 };
    const int LCP[12] = { 32, 64, 64, 160,128,128, 288,256,256, 288,256,256 };

    // ---- workspace carve-up ----
    char* ws = (char*)d_ws;
    size_t off = 0;
    auto carve = [&](size_t bytes) {
        void* p = (void*)(ws + off);
        off += (bytes + 255) & ~(size_t)255;
        return p;
    };
    float* XYZ1 = (float*)carve((size_t)B * N1 * 3 * 4);
    float* NX1  = (float*)carve((size_t)B * S1 * 3 * 4);
    float* NX2  = (float*)carve((size_t)B * S2 * 3 * 4);
    float* NX3  = (float*)carve((size_t)B * S3 * 3 * 4);
    int*   IDX1 = (int*)  carve((size_t)B * S1 * K1 * 4);
    int*   IDX2 = (int*)  carve((size_t)B * S2 * K2 * 4);
    int*   IDX3 = (int*)  carve((size_t)B * S3 * K3 * 4);
    float* POOL1= (float*)carve((size_t)B * S1 * 128 * 4);
    float* POOL2= (float*)carve((size_t)B * S2 * 256 * 4);
    float* POOL3= (float*)carve((size_t)B * S3 * 256 * 4);

    size_t woff[12], soff[12], wtot = 0, stot = 0;
    for (int L = 0; L < 12; ++L) {
        woff[L] = wtot; wtot += (size_t)LO[L] * LCP[L];
        soff[L] = stot; stot += (size_t)LO[L];
    }
    f16*   WF = (f16*)  carve(wtot * 2);
    float* SC = (float*)carve(stot * 4);
    float* BI = (float*)carve(stot * 4);

    f16* SLABG = (f16*)carve((size_t)131072 * 160 * 2);  // 40 MB grouped slab
    f16* SLABA = (f16*)carve((size_t)262144 * 128 * 2);  // 64 MB act ping
    f16* SLABB = (f16*)carve((size_t)262144 *  64 * 2);  // 32 MB act pong
    (void)ws_size; (void)in_sizes; (void)n_in; (void)out_size;

    // ---- weight / BN folding prep (all 12 layers) ----
    for (int L = 0; L < 12; ++L) {
        prep_weights_kernel<<<(LO[L] + 63) / 64, 64, 0, stream>>>(
            P(L,0), P(L,1), P(L,2), P(L,3), P(L,4), P(L,5),
            WF + woff[L], SC + soff[L], BI + soff[L], LO[L], LC[L], LCP[L]);
    }

    auto gemm = [&](const f16* X, int L, f16* Y, int M) {
        dim3 grid(M / 128, LO[L] / 64);
        mlp_wmma_kernel<<<grid, 256, 0, stream>>>(
            X, WF + woff[L], SC + soff[L], BI + soff[L], Y, M, LCP[L], LO[L]);
    };

    // ================= SA1: N=4096 -> S=1024, K=32, 3 -> 64,64,128 =========
    transpose_xyz_kernel<<<(B * N1 + 255) / 256, 256, 0, stream>>>(x, XYZ1, B, N1);
    fps_kernel<16><<<B, 256, 0, stream>>>(XYZ1, NX1, N1, S1);
    ball_query_kernel<<<(B * S1 + 255) / 256, 256, 0, stream>>>(
        XYZ1, NX1, IDX1, B, N1, S1, K1, 0.05f * 0.05f);
    int M1 = B * S1 * K1;  // 262144
    group_kernel<<<(M1 + 255) / 256, 256, 0, stream>>>(
        XYZ1, NX1, IDX1, (const float*)nullptr, SLABG, B, N1, S1, K1, 0, LCP[0]);
    gemm(SLABG, 0, SLABA, M1);
    gemm(SLABA, 1, SLABB, M1);
    gemm(SLABB, 2, SLABA, M1);
    maxpool_kernel<<<((size_t)B * S1 * 128 + 255) / 256, 256, 0, stream>>>(
        SLABA, POOL1, B * S1, K1, 128);

    // ================= SA2: N=1024 -> S=512, K=32, 131 -> 128,128,256 ======
    fps_kernel<4><<<B, 256, 0, stream>>>(NX1, NX2, S1, S2);
    ball_query_kernel<<<(B * S2 + 255) / 256, 256, 0, stream>>>(
        NX1, NX2, IDX2, B, S1, S2, K2, 0.1f * 0.1f);
    int M2 = B * S2 * K2;  // 131072
    group_kernel<<<(M2 + 255) / 256, 256, 0, stream>>>(
        NX1, NX2, IDX2, POOL1, SLABG, B, S1, S2, K2, 128, LCP[3]);
    gemm(SLABG, 3, SLABA, M2);
    gemm(SLABA, 4, SLABB, M2);
    gemm(SLABB, 5, SLABA, M2);
    maxpool_kernel<<<((size_t)B * S2 * 256 + 255) / 256, 256, 0, stream>>>(
        SLABA, POOL2, B * S2, K2, 256);

    // ================= SA3: N=512 -> S=128, K=64, 259 -> 256,256,256 =======
    fps_kernel<2><<<B, 256, 0, stream>>>(NX2, NX3, S2, S3);
    ball_query_kernel<<<(B * S3 + 255) / 256, 256, 0, stream>>>(
        NX2, NX3, IDX3, B, S2, S3, K3, 0.2f * 0.2f);
    int M3 = B * S3 * K3;  // 65536
    group_kernel<<<(M3 + 255) / 256, 256, 0, stream>>>(
        NX2, NX3, IDX3, POOL2, SLABG, B, S2, S3, K3, 256, LCP[6]);
    gemm(SLABG, 6, SLABA, M3);
    gemm(SLABA, 7, SLABB, M3);
    gemm(SLABB, 8, SLABA, M3);
    maxpool_kernel<<<((size_t)B * S3 * 256 + 255) / 256, 256, 0, stream>>>(
        SLABA, POOL3, B * S3, K3, 256);

    // ================= SA4: group_all over 128 pts, 259 -> 256,256,256 =====
    int M4 = B * N4;  // 1024
    group_all_kernel<<<(M4 + 255) / 256, 256, 0, stream>>>(
        NX3, POOL3, SLABG, B, N4, 256, LCP[9]);
    gemm(SLABG,  9, SLABA, M4);
    gemm(SLABA, 10, SLABB, M4);
    gemm(SLABB, 11, SLABA, M4);
    maxpool_kernel<<<((size_t)B * 256 + 255) / 256, 256, 0, stream>>>(
        SLABA, (float*)d_out, B, N4, 256);
}
